// RNN_Navi_38568806318894
// MI455X (gfx1250) — compile-verified, matrix operand order checked by reading, not run on hardware
//
#include <hip/hip_runtime.h>
#include <hip/hip_bf16.h>

// ---------------------------------------------------------------------------
// LIF RNN:  B=64, T=256, I=3, N=2048, OUT=2, alpha=0.1
//   i_t = x[:,t,:] @ W_i ;  r0 = x0 @ W_x0
//   u_t = i_t + r_{t-1} + b ; a_t = relu(u_t)
//   r_t = 0.9 r_{t-1} + 0.1 (a_t @ W_r) ; f_t = a_t @ W_f
// Outputs: f [B,T,OUT] then u [B,T,N], concatenated flat in d_out.
//
// Strategy (MI455X, gfx1250, wave32):
//  * persistent kernel, 128 WGs (one per 16-col tile of N), 4 waves (M-tiles of B)
//  * recurrent state r kept in WMMA f32 accumulator layout registers
//  * per-step device barrier: agent-scope atomic counter (release/acquire)
//  * a_t stored as f16, double-buffered by step parity
//  * a@W_r via v_wmma_f32_16x16x32_f16, K-loop of 64, 2 accumulators
//  * W_r pre-packed into per-lane B-operand layout (contiguous 32B/lane/chunk)
//  * readout: deterministic per-tile partials + final reduction kernel
// ---------------------------------------------------------------------------

typedef _Float16 v16h __attribute__((ext_vector_type(16)));
typedef _Float16 v8h  __attribute__((ext_vector_type(8)));
typedef float    v8f  __attribute__((ext_vector_type(8)));

#define B_    64
#define T_    256
#define I_    3
#define N_    2048
#define OUT_  2
#define ALPHA 0.1f
#define NT    (N_ / 16)   // 128 column tiles == number of workgroups
#define KC    (N_ / 32)   // 64 K-chunks of 32

// ---- workspace layout (bytes) ----
#define WS_COUNTER 0
#define WS_ABUF    1024                                   // f16 [2][B_][N_]
#define WS_PACKED  (WS_ABUF + 2 * B_ * N_ * 2)            // f16 [NT][KC][32][16]
#define WS_PART    (WS_PACKED + (size_t)N_ * N_ * 2)      // f32 [T_][NT][B_][OUT_]

__global__ void init_counter(unsigned* c) { *c = 0u; }

// Pack W_r (f32 row-major [K=N_][N=N_]) into WMMA B-operand tiles, f16.
// Tile (nt, kc): lane l holds column n = nt*16 + (l&15); element j (0..15)
// holds K = kc*32 + ((l>=16)?8:0) + (j<8 ? j : j+8)   (matches 16-bit A/B
// VGPR layout: VGPR v = j/2, half = j&1).
__global__ void pack_wr(const float* __restrict__ Wr, _Float16* __restrict__ pk) {
  int idx = blockIdx.x * blockDim.x + threadIdx.x;
  if (idx >= N_ * N_) return;
  int j  = idx & 15;
  int l  = (idx >> 4) & 31;
  int kc = (idx >> 9) & (KC - 1);
  int nt = idx >> 15;
  int n  = nt * 16 + (l & 15);
  int k  = kc * 32 + ((l & 16) ? 8 : 0) + ((j < 8) ? j : (j + 8));
  pk[idx] = (_Float16)Wr[k * N_ + n];
}

__device__ __forceinline__ v16h merge16(v8h lo, v8h hi) {
  return __builtin_shufflevector(lo, hi, 0,1,2,3,4,5,6,7,8,9,10,11,12,13,14,15);
}

__global__ void __launch_bounds__(128)
rnn_persist(const float* __restrict__ x,   const float* __restrict__ x0,
            const float* __restrict__ Wi,  const float* __restrict__ Wx0,
            const float* __restrict__ Wf,  const float* __restrict__ bias,
            const _Float16* __restrict__ wr_pk,
            _Float16* __restrict__ a_buf,          // 2 * B_ * N_ halves
            float* __restrict__ part,              // T_*NT*B_*OUT_
            float* __restrict__ act,               // B_*T_*N_
            unsigned* __restrict__ counter) {
  const int nt   = blockIdx.x;           // column tile of N
  const int wave = threadIdx.x >> 5;     // M-tile of B (0..3)
  const int lane = threadIdx.x & 31;
  const int l16  = lane & 15;
  const int hs   = lane >> 4;            // 0 (lanes 0-15) / 1 (lanes 16-31)

  const int n = nt * 16 + l16;           // this lane's column in C/D layout
  // Per-lane constants (column-indexed weights)
  const float wi0 = Wi[0 * N_ + n], wi1 = Wi[1 * N_ + n], wi2 = Wi[2 * N_ + n];
  const float wx0 = Wx0[0 * N_ + n], wx1 = Wx0[1 * N_ + n], wx2 = Wx0[2 * N_ + n];
  const float bb  = bias[n];
  const float wf0 = Wf[n * OUT_ + 0], wf1 = Wf[n * OUT_ + 1];

  // r in C/D accumulator layout: element v -> row M = v + 8*hs, col = l16
  float r[8];
#pragma unroll
  for (int v = 0; v < 8; ++v) {
    const int b = wave * 16 + v + 8 * hs;
    r[v] = x0[b * I_ + 0] * wx0 + x0[b * I_ + 1] * wx1 + x0[b * I_ + 2] * wx2;
  }

  const _Float16* wr_base = wr_pk + (size_t)nt * KC * 512 + lane * 16;

  for (int t = 0; t < T_; ++t) {
    _Float16* abuf = a_buf + (size_t)(t & 1) * (B_ * N_);

    // ---- stage A: membrane potential, act store, a store, readout partials
    float pf0[8], pf1[8];
#pragma unroll
    for (int v = 0; v < 8; ++v) {
      const int b = wave * 16 + v + 8 * hs;
      const float* xp = x + ((size_t)b * T_ + t) * I_;
      const float u = r[v] + xp[0] * wi0 + xp[1] * wi1 + xp[2] * wi2 + bb;
      act[((size_t)b * T_ + t) * N_ + n] = u;
      const float a = u > 0.0f ? u : 0.0f;
      abuf[b * N_ + n] = (_Float16)a;
      pf0[v] = a * wf0;
      pf1[v] = a * wf1;
    }
    // deterministic 16-lane reductions (stay inside each half of the wave)
#pragma unroll
    for (int v = 0; v < 8; ++v) {
#pragma unroll
      for (int m = 1; m < 16; m <<= 1) {
        pf0[v] += __shfl_xor(pf0[v], m);
        pf1[v] += __shfl_xor(pf1[v], m);
      }
    }
    if (l16 == 0) {
#pragma unroll
      for (int v = 0; v < 8; ++v) {
        const int b = wave * 16 + v + 8 * hs;
        float2* p = (float2*)(part + (((size_t)t * NT + nt) * B_ + b) * OUT_);
        *p = make_float2(pf0[v], pf1[v]);
      }
    }

    // ---- device-wide barrier (monotone counter; release arrive, acquire spin)
    __syncthreads();
    if (threadIdx.x == 0) {
      __hip_atomic_fetch_add(counter, 1u, __ATOMIC_RELEASE, __HIP_MEMORY_SCOPE_AGENT);
      const unsigned target = (unsigned)(t + 1) * NT;
      while (__hip_atomic_load(counter, __ATOMIC_ACQUIRE, __HIP_MEMORY_SCOPE_AGENT) < target)
        __builtin_amdgcn_s_sleep(1);
    }
    __syncthreads();

    // ---- stage B: acc = a_t @ W_r[:, tile] via f16 WMMA, f32 accumulate
    // A layout: lane holds row M = l16 of this wave's M-tile; K offset +8*hs.
    const _Float16* arow = abuf + (size_t)(wave * 16 + l16) * N_ + 8 * hs;
    v8f acc0 = {0.f, 0.f, 0.f, 0.f, 0.f, 0.f, 0.f, 0.f};
    v8f acc1 = {0.f, 0.f, 0.f, 0.f, 0.f, 0.f, 0.f, 0.f};
#pragma unroll 4
    for (int kc = 0; kc < KC; kc += 2) {
      const v16h A0 = merge16(*(const v8h*)(arow + kc * 32),
                              *(const v8h*)(arow + kc * 32 + 16));
      const v16h B0 = merge16(*(const v8h*)(wr_base + (size_t)kc * 512),
                              *(const v8h*)(wr_base + (size_t)kc * 512 + 8));
      const v16h A1 = merge16(*(const v8h*)(arow + (kc + 1) * 32),
                              *(const v8h*)(arow + (kc + 1) * 32 + 16));
      const v16h B1 = merge16(*(const v8h*)(wr_base + (size_t)(kc + 1) * 512),
                              *(const v8h*)(wr_base + (size_t)(kc + 1) * 512 + 8));
      acc0 = __builtin_amdgcn_wmma_f32_16x16x32_f16(false, A0, false, B0,
                                                    (short)0, acc0, false, false);
      acc1 = __builtin_amdgcn_wmma_f32_16x16x32_f16(false, A1, false, B1,
                                                    (short)0, acc1, false, false);
    }
    // leaky update (f32)
#pragma unroll
    for (int v = 0; v < 8; ++v)
      r[v] = (1.0f - ALPHA) * r[v] + ALPHA * (acc0[v] + acc1[v]);
  }
}

// out[b,t,o] = sum_nt part[t][nt][b][o]  (deterministic fixed-order sum)
__global__ void reduce_f(const float* __restrict__ part, float* __restrict__ out) {
  const int id = blockIdx.x * blockDim.x + threadIdx.x;
  if (id >= B_ * T_ * OUT_) return;
  const int o = id & (OUT_ - 1);
  const int bt = id / OUT_;
  const int t = bt % T_;
  const int b = bt / T_;
  float s = 0.0f;
  for (int nt = 0; nt < NT; ++nt)
    s += part[(((size_t)t * NT + nt) * B_ + b) * OUT_ + o];
  out[id] = s;
}

extern "C" void kernel_launch(void* const* d_in, const int* in_sizes, int n_in,
                              void* d_out, int out_size, void* d_ws, size_t ws_size,
                              hipStream_t stream) {
  (void)in_sizes; (void)n_in; (void)out_size; (void)ws_size;
  const float* x   = (const float*)d_in[0];  // [B,T,I]
  const float* x0  = (const float*)d_in[1];  // [B,I]
  const float* Wi  = (const float*)d_in[2];  // [I,N]
  const float* Wx0 = (const float*)d_in[3];  // [I,N]
  const float* Wr  = (const float*)d_in[4];  // [N,N]
  const float* Wf  = (const float*)d_in[5];  // [N,OUT]
  const float* b   = (const float*)d_in[6];  // [N]

  float* out = (float*)d_out;                // [B,T,OUT]
  float* act = out + (size_t)B_ * T_ * OUT_; // [B,T,N]

  char* ws = (char*)d_ws;
  unsigned*  counter = (unsigned*)(ws + WS_COUNTER);
  _Float16*  abuf    = (_Float16*)(ws + WS_ABUF);
  _Float16*  wpk     = (_Float16*)(ws + WS_PACKED);
  float*     part    = (float*)(ws + WS_PART);

  init_counter<<<1, 1, 0, stream>>>(counter);
  pack_wr<<<(N_ * N_ + 255) / 256, 256, 0, stream>>>(Wr, wpk);
  // 128 WGs x 128 threads: well under residency limits -> spin barrier is safe
  rnn_persist<<<NT, 128, 0, stream>>>(x, x0, Wi, Wx0, Wf, b, wpk, abuf, part,
                                      act, counter);
  reduce_f<<<(B_ * T_ * OUT_ + 255) / 256, 256, 0, stream>>>(part, out);
}
